// SimpleLSTM_5763846111454
// MI455X (gfx1250) — matrix operand order repaired and checked by reading, not compiled, not run on hardware
//
#include <hip/hip_runtime.h>
#include <hip/hip_bf16.h>
#include <stdint.h>

// ---------------- problem constants ----------------
#define HIDDEN   256
#define BATCH    64
#define SEQ      2048
#define GATES    1024          // 4*HIDDEN
#define GST      1040          // padded bf16 row stride for activated-gates LDS
#define HSTRIDE  264           // padded bf16 row stride for h LDS (132 dwords)
#define NTHREADS 512           // 16 waves

// LDS layout (bytes)
#define BLDS_BYTES   262144                        // 256KB: W_hh k-tiles 4..7
#define GATES_OFF    BLDS_BYTES                    // 16*1040 bf16 = 33280B
#define HBUF_OFF     (GATES_OFF + 16*GST*2)        // 16*264 bf16 = 8448B
#define XBUF_OFF     (HBUF_OFF + 16*HSTRIDE*2)     // 2*16 f32
#define WIH_OFF      (XBUF_OFF + 32*4)             // 1024 f32 = 4KB
#define GB_OFF       (WIH_OFF + 1024*4)            // 1024 f32 = 4KB
#define WFC_OFF      (GB_OFF + 1024*4)             // 256 f32 = 1KB
#define SMEM_BYTES   (WFC_OFF + 256*4)

typedef __bf16 bf16_t;
typedef __attribute__((ext_vector_type(16))) __bf16 v16bf;
typedef __attribute__((ext_vector_type(8)))  float  v8f;

__device__ __forceinline__ uint32_t f32_to_bf16_bits(float f) {
  union { float f; uint32_t u; } v; v.f = f;
  uint32_t u = v.u;
  uint32_t r = u + 0x7FFFu + ((u >> 16) & 1u);   // round-to-nearest-even
  return r >> 16;
}
__device__ __forceinline__ uint32_t pack_bf16(float lo, float hi) {
  return f32_to_bf16_bits(lo) | (f32_to_bf16_bits(hi) << 16);
}
__device__ __forceinline__ float bf16_bits_to_f32(uint32_t b) {
  union { uint32_t u; float f; } v; v.u = b << 16; return v.f;
}
__device__ __forceinline__ float fsigmoid(float x) {
  return 1.0f / (1.0f + __expf(-x));
}
__device__ __forceinline__ float ftanh(float x) {
  return 1.0f - 2.0f / (__expf(2.0f * x) + 1.0f);
}

// ---------------------------------------------------------------------------
// Prep kernel: swizzle W_hh (f32 [1024,256], row = gate col) into WMMA
// B-matrix (32x16 bf16, K x N) fragments, fragment-major in d_ws:
//   frag = nt*8 + k (nt: N-tile of 16, 0..63; k: K-tile of 32, 0..7)
//   per frag: 32 lanes x 8 dwords (lane-contiguous 32B).
// B layout: lane holds N col (lane&15); dword v holds K = 8*(lane>>4) + 2v (+8 if v>=4)
// (lo 16b) and K+1 (hi 16b).  B[k][n] = W_hh[n*256+k].
// ---------------------------------------------------------------------------
__global__ void lstm_prep_kernel(const float* __restrict__ Whh,
                                 uint32_t* __restrict__ bw) {
  int id   = blockIdx.x * blockDim.x + threadIdx.x;  // 131072 total
  int dw   = id & 7;
  int lane = (id >> 3) & 31;
  int frag = id >> 8;
  int k    = frag & 7;
  int nt   = frag >> 3;
  int K    = k * 32 + 8 * (lane >> 4) + 2 * dw + ((dw >> 2) << 3);
  int n0   = nt * 16 + (lane & 15);
  bw[id]   = pack_bf16(Whh[n0 * HIDDEN + K], Whh[n0 * HIDDEN + K + 1]);
}

// ---------------------------------------------------------------------------
// Main persistent kernel: 4 blocks x 512 threads (16 waves); block owns 16
// batch rows; wave w owns gate columns [64w, 64w+64) (4 N-tiles).
// W_hh residency: k-tiles 0..3 in VGPRs (16 frags = 128/wave), k-tiles 4..7
// in LDS (256KB). Steady state touches no global memory for weights.
// ---------------------------------------------------------------------------
__global__ void __launch_bounds__(NTHREADS)
lstm_main_kernel(const float* __restrict__ x,
                 const float* __restrict__ Wih,
                 const float* __restrict__ bih,
                 const float* __restrict__ bhh,
                 const float* __restrict__ Wfc,
                 const float* __restrict__ bfc,
                 const uint32_t* __restrict__ bw,
                 float* __restrict__ out) {
  extern __shared__ char smem[];
  uint32_t* bLds   = (uint32_t*)smem;                   // 256KB B k=4..7
  uint16_t* gatesB = (uint16_t*)(smem + GATES_OFF);     // activated gates bf16
  bf16_t*   hbuf   = (bf16_t*)(smem + HBUF_OFF);        // h state bf16
  float*    xbuf   = (float*)(smem + XBUF_OFF);         // 2x16 double buffer
  float*    wihL   = (float*)(smem + WIH_OFF);          // W_ih per gate col
  float*    gbL    = (float*)(smem + GB_OFF);           // b_ih+b_hh per col
  float*    wfcL   = (float*)(smem + WFC_OFF);          // W_fc per h col

  const int tid  = threadIdx.x;
  const int wave = tid >> 5;
  const int lane = tid & 31;
  const int r0   = blockIdx.x * 16;

  // ---- GEMM-role constants: wave owns n-tiles 4w..4w+3 ----
  const int l15  = lane & 15;
  const int hi   = lane >> 4;
  const int cbase = wave * 64 + l15;              // + 16*n for n-tile n
  const bool isG  = ((wave >> 2) == 2);           // waves 8..11 = g gate (tanh)
  const int nt0   = wave * 4;

  // ---- elementwise-role constants: wave == row, lane owns 8 cols ----
  const int er = wave;                            // one wave per batch row
  const int ej = lane * 8;
  const float bfc0 = bfc[0];

  float creg[8] = {0.f,0.f,0.f,0.f,0.f,0.f,0.f,0.f};

  const uint4* bwq = (const uint4*)bw;

  // ---- stage B k-tiles 4..7 into LDS (16384 uint4 = 256KB) ----
  {
    uint4* bLq = (uint4*)bLds;
    for (int i = tid; i < 16384; i += NTHREADS) {
      int frag2 = i >> 6;                 // nt*4 + (k-4)
      int rem   = i & 63;
      int nt    = frag2 >> 2;
      int k     = (frag2 & 3) + 4;
      bLq[i] = bwq[(nt * 8 + k) * 64 + rem];
    }
  }
  // ---- pin B k-tiles 0..3 in registers (16 frags = 128 VGPRs) ----
  union Frag { v16bf v; uint4 q[2]; };
  Frag Br[4][4];                                  // [n][k]
  #pragma unroll
  for (int n = 0; n < 4; ++n) {
    #pragma unroll
    for (int k = 0; k < 4; ++k) {
      int f = (((nt0 + n) * 8 + k) * 32 + lane) * 2;
      Br[n][k].q[0] = bwq[f];
      Br[n][k].q[1] = bwq[f + 1];
    }
  }

  // constant tables, zero h buffer, x for t=0
  for (int i = tid; i < GATES; i += NTHREADS) {
    wihL[i] = Wih[i];
    gbL[i]  = bih[i] + bhh[i];
  }
  if (tid < HIDDEN) wfcL[tid] = Wfc[tid];
  for (int i = tid; i < 16 * HSTRIDE; i += NTHREADS) hbuf[i] = (bf16_t)0.0f;
  if (tid < 16) xbuf[tid] = x[(r0 + tid) * SEQ + 0];
  __syncthreads();

  const uint4* ldsA = (const uint4*)hbuf;
  const uint4* bLq4 = (const uint4*)bLds;

  for (int t = 0; t < SEQ; ++t) {
    // ============== GEMM phase: 16 rows x 64 gate cols per wave ==========
    v8f acc[4];
    #pragma unroll
    for (int n = 0; n < 4; ++n) acc[n] = (v8f){};
    #pragma unroll
    for (int k = 0; k < 4; ++k) {         // register-resident B
      Frag A;
      int ai = l15 * 33 + k * 4 + hi;     // A 16x32 bf16, 2x ds_b128
      A.q[0] = ldsA[ai];
      A.q[1] = ldsA[ai + 2];
      #pragma unroll
      for (int n = 0; n < 4; ++n)
        acc[n] = __builtin_amdgcn_wmma_f32_16x16x32_bf16(
            false, A.v, false, Br[n][k].v, (short)0, acc[n], false, false);
    }
    #pragma unroll
    for (int k = 4; k < 8; ++k) {         // LDS-resident B
      Frag A;
      int ai = l15 * 33 + k * 4 + hi;
      A.q[0] = ldsA[ai];
      A.q[1] = ldsA[ai + 2];
      #pragma unroll
      for (int n = 0; n < 4; ++n) {
        Frag B;
        int f = (((nt0 + n) * 4 + (k - 4)) * 32 + lane) * 2;
        B.q[0] = bLq4[f];
        B.q[1] = bLq4[f + 1];
        acc[n] = __builtin_amdgcn_wmma_f32_16x16x32_bf16(
            false, A.v, false, B.v, (short)0, acc[n], false, false);
      }
    }
    // add input term, activate, store activated gates (bf16) to LDS.
    // C tile: lane = N col, dword v = row v (lanes 0-15) / v+8 (lanes 16-31).
    {
      const int xp = (t & 1) * 16;
      #pragma unroll
      for (int n = 0; n < 4; ++n) {
        const int col = cbase + 16 * n;
        const float wih = wihL[col];
        const float gb  = gbL[col];
        #pragma unroll
        for (int v = 0; v < 8; ++v) {
          int row = v + hi * 8;
          float g = acc[n][v] + xbuf[xp + row] * wih + gb;
          float a = isG ? ftanh(g) : fsigmoid(g);
          gatesB[row * GST + col] = (uint16_t)f32_to_bf16_bits(a);
        }
      }
    }
    __syncthreads();

    // ============== elementwise phase: wave er handles row er ==============
    if (tid < 16) {   // prefetch x for t+1 (double-buffered)
      int tn = (t + 1 < SEQ) ? (t + 1) : (SEQ - 1);
      xbuf[((t + 1) & 1) * 16 + tid] = x[(r0 + tid) * SEQ + tn];
    }
    const int gbase = er * GST + ej;      // bf16 units, 16B aligned
    uint4 ui = *(const uint4*)&gatesB[gbase];
    uint4 uf = *(const uint4*)&gatesB[gbase + 256];
    uint4 ug = *(const uint4*)&gatesB[gbase + 512];
    uint4 uo = *(const uint4*)&gatesB[gbase + 768];
    const uint32_t* wi4 = (const uint32_t*)&ui;
    const uint32_t* wf4 = (const uint32_t*)&uf;
    const uint32_t* wg4 = (const uint32_t*)&ug;
    const uint32_t* wo4 = (const uint32_t*)&uo;
    float yp = 0.0f;
    float hn[8];
    #pragma unroll
    for (int j = 0; j < 8; ++j) {
      int d = j >> 1, sh = (j & 1) * 16;
      float si = bf16_bits_to_f32((wi4[d] >> sh) & 0xFFFFu);
      float sf = bf16_bits_to_f32((wf4[d] >> sh) & 0xFFFFu);
      float tg = bf16_bits_to_f32((wg4[d] >> sh) & 0xFFFFu);
      float so = bf16_bits_to_f32((wo4[d] >> sh) & 0xFFFFu);
      float cn = sf * creg[j] + si * tg;
      hn[j] = so * ftanh(cn);
      creg[j] = cn;
      yp += wfcL[ej + j] * hn[j];
    }
    // h back to LDS (bf16) for next step's A matrix (16B aligned)
    {
      uint4 p;
      p.x = pack_bf16(hn[0], hn[1]);
      p.y = pack_bf16(hn[2], hn[3]);
      p.z = pack_bf16(hn[4], hn[5]);
      p.w = pack_bf16(hn[6], hn[7]);
      *(uint4*)&hbuf[er * HSTRIDE + ej] = p;
    }
    // fused FC head: whole row lives in this wave -> finish in-wave
    #pragma unroll
    for (int off = 16; off > 0; off >>= 1) yp += __shfl_xor(yp, off, 32);
    if (lane == 0) out[(r0 + er) * SEQ + t] = yp + bfc0;
    __syncthreads();
  }

  // ---- final h_n (from LDS h state), c_n (full f32 registers) ----
  const int yN = BATCH * SEQ;
  {
    uint4 p = *(const uint4*)&hbuf[er * HSTRIDE + ej];
    const uint32_t* pw = (const uint32_t*)&p;
    #pragma unroll
    for (int j = 0; j < 8; ++j) {
      int d = j >> 1, sh = (j & 1) * 16;
      out[yN + (r0 + er) * HIDDEN + ej + j] =
          bf16_bits_to_f32((pw[d] >> sh) & 0xFFFFu);
    }
  }
  #pragma unroll
  for (int j = 0; j < 8; ++j)
    out[yN + BATCH * HIDDEN + (r0 + er) * HIDDEN + ej + j] = creg[j];
}

// ---------------------------------------------------------------------------
extern "C" void kernel_launch(void* const* d_in, const int* in_sizes, int n_in,
                              void* d_out, int out_size, void* d_ws, size_t ws_size,
                              hipStream_t stream) {
  (void)in_sizes; (void)n_in; (void)out_size; (void)ws_size;
  const float* x   = (const float*)d_in[0];
  const float* Wih = (const float*)d_in[1];
  const float* Whh = (const float*)d_in[2];
  const float* bih = (const float*)d_in[3];
  const float* bhh = (const float*)d_in[4];
  const float* Wfc = (const float*)d_in[5];
  const float* bfc = (const float*)d_in[6];
  float* out       = (float*)d_out;
  uint32_t* bw     = (uint32_t*)d_ws;      // 512KB swizzled bf16 W_hh fragments

  lstm_prep_kernel<<<512, 256, 0, stream>>>(Whh, bw);
  lstm_main_kernel<<<4, NTHREADS, SMEM_BYTES, stream>>>(x, Wih, bih, bhh,
                                                        Wfc, bfc, bw, out);
}
